// PSMM_17188459119361
// MI455X (gfx1250) — matrix operand order repaired and checked by reading, not compiled, not run on hardware
//
#include <hip/hip_runtime.h>
#include <hip/hip_bf16.h>

// ---------------------------------------------------------------------------
// Types for CDNA5 WMMA (gfx1250, wave32)
// ---------------------------------------------------------------------------
typedef unsigned short u16;
typedef __attribute__((ext_vector_type(16))) __bf16 bf16x16;
typedef __attribute__((ext_vector_type(8)))  __bf16 bf16x8;
typedef __attribute__((ext_vector_type(8)))  float  f32x8;

#define L_SEQ 192
#define BATCH 32
#define VOCAB 8000
#define HID   1024
#define GATES 4096   // 4*HID
#define ROWS  6144   // L_SEQ*BATCH
#define LSTM_WGS 32
#define LDS_ROW 1032                 // padded W_hh row stride (elements) -> bank-conflict-free
#define LSTM_LDS_BYTES (128 * LDS_ROW * 2 + BATCH * 128 * 4)   // 264192 + 16384 = 280576

__device__ __forceinline__ u16 f2bf(float f) {
  unsigned u = __float_as_uint(f);
  u += 0x7fffu + ((u >> 16) & 1u);   // round-to-nearest-even
  return (u16)(u >> 16);
}

// WMMA wrapper: D = A(16x32 bf16) * B(32x16 bf16) + C(16x16 f32)
__device__ __forceinline__ f32x8 wmma_bf16(bf16x16 a, bf16x16 b, f32x8 c) {
  return __builtin_amdgcn_wmma_f32_16x16x32_bf16(
      /*neg_a=*/false, a, /*neg_b=*/false, b,
      /*c_mod=*/(short)0, c, /*reuse_a=*/false, /*reuse_b=*/false);
}

union Frag { bf16x16 v; bf16x8 h[2]; };

// ---------------------------------------------------------------------------
// Conversion / gather helpers
// ---------------------------------------------------------------------------
__global__ void __launch_bounds__(256) f32_to_bf16_kernel(const float* __restrict__ src,
                                                          u16* __restrict__ dst, int n) {
  int i = blockIdx.x * 256 + threadIdx.x;
  if (i < n) dst[i] = f2bf(src[i]);
}

__global__ void __launch_bounds__(256) embed_gather_kernel(const int* __restrict__ tokens,
                                                           const float* __restrict__ table,
                                                           u16* __restrict__ embB) {
  int r = blockIdx.x;                 // row = t*B + b  (matches tokens flat layout)
  int tok = tokens[r];
  const float* src = table + (size_t)tok * HID;
  u16* dst = embB + (size_t)r * HID;
  for (int i = threadIdx.x; i < HID; i += 256) dst[i] = f2bf(src[i]);
}

__global__ void __launch_bounds__(256) bias_combine_kernel(const float* __restrict__ a,
                                                           const float* __restrict__ b,
                                                           float* __restrict__ o, int n) {
  int i = blockIdx.x * 256 + threadIdx.x;
  if (i < n) o[i] = a[i] + b[i];
}

// ---------------------------------------------------------------------------
// bf16 WMMA GEMM:  C[m,n] = act( sum_k A[m,k]*B[n,k] + bias[n] )
// A: MxK row-major bf16, B: NxK row-major bf16 (weight rows = output cols).
// WG = 256 threads = 8 waves; macro tile 64(M) x 64(N); each wave computes a
// 16x32 strip (two 16x16 tiles sharing one A fragment) -> 2 WMMA per 4 loads.
// All operands are L2-resident (192 MB L2 covers every matrix), so no LDS
// staging is required to approach the WMMA issue rate.
// ---------------------------------------------------------------------------
template <int ACT>  // 0 = none, 1 = tanh
__global__ void __launch_bounds__(256) gemm_bf16_kernel(const u16* __restrict__ A,
                                                        const u16* __restrict__ B,
                                                        const float* __restrict__ bias,
                                                        float* __restrict__ C,
                                                        int M, int N, int K) {
  const int wave = threadIdx.x >> 5;
  const int lane = threadIdx.x & 31;
  const int nl   = lane & 15;
  const int half = lane >> 4;

  const int m0 = blockIdx.y * 64 + (wave >> 1) * 16;
  const int n0 = blockIdx.x * 64 + (wave & 1) * 32;

  // A fragment: lane holds row m0+nl, 16 contiguous K at half*16
  const u16* a_ptr  = A + (size_t)(m0 + nl) * K + half * 16;
  // B fragments: lane holds weight row (output col), 16 contiguous K at half*16
  const u16* b_ptr0 = B + (size_t)(n0 + nl) * K + half * 16;
  const u16* b_ptr1 = B + (size_t)(n0 + 16 + nl) * K + half * 16;

  f32x8 acc0 = {}, acc1 = {};
  for (int kb = 0; kb < K; kb += 32) {
    __builtin_prefetch(a_ptr + kb + 256, 0, 0);
    __builtin_prefetch(b_ptr0 + kb + 256, 0, 0);
    __builtin_prefetch(b_ptr1 + kb + 256, 0, 0);
    Frag ua;
    ua.h[0] = *reinterpret_cast<const bf16x8*>(a_ptr + kb);
    ua.h[1] = *reinterpret_cast<const bf16x8*>(a_ptr + kb + 8);
    bf16x16 b0 = *reinterpret_cast<const bf16x16*>(b_ptr0 + kb);
    bf16x16 b1 = *reinterpret_cast<const bf16x16*>(b_ptr1 + kb);
    acc0 = wmma_bf16(ua.v, b0, acc0);
    acc1 = wmma_bf16(ua.v, b1, acc1);
  }

  const int mrow = m0 + half * 8;
  const int na = n0 + nl;
  const int nb = n0 + 16 + nl;
  const float bna = bias[na];
  const float bnb = bias[nb];
  for (int r = 0; r < 8; ++r) {
    float va = acc0[r] + bna;
    float vb = acc1[r] + bnb;
    if (ACT == 1) { va = tanhf(va); vb = tanhf(vb); }
    C[(size_t)(mrow + r) * N + na] = va;
    C[(size_t)(mrow + r) * N + nb] = vb;
  }
}

// ---------------------------------------------------------------------------
// Cooperative LSTM: 32 workgroups, each owns hidden chunk [wg*32, wg*32+32).
// The WG's 128 W_hh rows (256 KB bf16) are staged ONCE into CDNA5's 320 KB
// LDS (padded stride to spread banks); the 192-step recurrence then reads
// B-fragments from LDS, eliminating the scratch spill/reload of weights.
// Per step: 16 WMMA tiles (8 waves x 2, shared A fragment), pointwise gates,
// two device-scope barriers (hB read-complete / write-complete).
// ---------------------------------------------------------------------------
__device__ __forceinline__ void grid_barrier(int* count, int* flag, int gen) {
  __threadfence();
  __syncthreads();
  if (threadIdx.x == 0) {
    int prev = __hip_atomic_fetch_add(count, 1, __ATOMIC_ACQ_REL, __HIP_MEMORY_SCOPE_AGENT);
    if (prev == LSTM_WGS - 1) {
      __hip_atomic_store(count, 0, __ATOMIC_RELAXED, __HIP_MEMORY_SCOPE_AGENT);
      __hip_atomic_store(flag, gen + 1, __ATOMIC_RELEASE, __HIP_MEMORY_SCOPE_AGENT);
    } else {
      while (__hip_atomic_load(flag, __ATOMIC_ACQUIRE, __HIP_MEMORY_SCOPE_AGENT) <= gen) {
        __builtin_amdgcn_s_sleep(1);
      }
    }
  }
  __syncthreads();
}

__global__ void __launch_bounds__(256, 1) lstm_kernel(const float* __restrict__ gates_x,
                                                      const u16* __restrict__ WhhB,
                                                      float* __restrict__ hiddens,
                                                      u16* __restrict__ hiddensB,
                                                      u16* __restrict__ hB,
                                                      float* __restrict__ cbuf,
                                                      int* __restrict__ sync_count,
                                                      int* __restrict__ sync_flag) {
  extern __shared__ char smem[];
  u16*   wlds = (u16*)smem;                                 // 128 x LDS_ROW bf16
  float* gl   = (float*)(smem + 128 * LDS_ROW * 2);         // [batch][4*32 gate vals]

  const int wg   = blockIdx.x;        // hidden chunk
  const int wave = threadIdx.x >> 5;
  const int lane = threadIdx.x & 31;
  const int nl   = lane & 15;
  const int half = lane >> 4;

  // ---- stage this WG's 128 W_hh rows into LDS (once) --------------------
  for (int ch = threadIdx.x; ch < 128 * (HID / 8); ch += 256) {
    int lr = ch >> 7;          // local row 0..127  (gi*32 + jj)
    int cc = ch & 127;         // 16-byte chunk within row
    int gr = (lr >> 5) * HID + wg * 32 + (lr & 31);   // global W_hh row
    *reinterpret_cast<int4*>(wlds + (size_t)lr * LDS_ROW + cc * 8) =
        *reinterpret_cast<const int4*>(WhhB + (size_t)gr * HID + cc * 8);
  }
  __syncthreads();

  const int gi = wave >> 1;            // gate type 0..3 (i,f,g,o)
  const int m0 = (wave & 1) * 16;      // batch-row half

  const u16* a_row = hB + (size_t)(m0 + nl) * HID + half * 16;
  const u16* bl0 = wlds + (size_t)(gi * 32 + nl) * LDS_ROW + half * 16;
  const u16* bl1 = wlds + (size_t)(gi * 32 + 16 + nl) * LDS_ROW + half * 16;

  for (int t = 0; t < L_SEQ; ++t) {
    f32x8 acc0 = {}, acc1 = {};
#pragma unroll 8
    for (int kb = 0; kb < HID; kb += 32) {
      Frag ua, ub0, ub1;
      ua.h[0]  = *reinterpret_cast<const bf16x8*>(a_row + kb);
      ua.h[1]  = *reinterpret_cast<const bf16x8*>(a_row + kb + 8);
      ub0.h[0] = *reinterpret_cast<const bf16x8*>(bl0 + kb);
      ub0.h[1] = *reinterpret_cast<const bf16x8*>(bl0 + kb + 8);
      ub1.h[0] = *reinterpret_cast<const bf16x8*>(bl1 + kb);
      ub1.h[1] = *reinterpret_cast<const bf16x8*>(bl1 + kb + 8);
      acc0 = wmma_bf16(ua.v, ub0.v, acc0);
      acc1 = wmma_bf16(ua.v, ub1.v, acc1);
    }
    const int lc = gi * 32;
    for (int r = 0; r < 8; ++r) {
      int brow = m0 + half * 8 + r;
      gl[brow * 128 + lc + nl]      = acc0[r];
      gl[brow * 128 + lc + 16 + nl] = acc1[r];
    }
    __syncthreads();
    grid_barrier(sync_count, sync_flag, 2 * t);      // all WGPs done reading hB

    for (int p = threadIdx.x; p < BATCH * 32; p += 256) {
      int b  = p >> 5;
      int jj = p & 31;
      int j  = wg * 32 + jj;
      const float* gx = gates_x + (size_t)(t * BATCH + b) * GATES;
      float zi = gl[b * 128 + jj]       + gx[j];
      float zf = gl[b * 128 + 32 + jj]  + gx[HID + j];
      float zg = gl[b * 128 + 64 + jj]  + gx[2 * HID + j];
      float zo = gl[b * 128 + 96 + jj]  + gx[3 * HID + j];
      float si = 1.f / (1.f + expf(-zi));
      float sf = 1.f / (1.f + expf(-zf));
      float so = 1.f / (1.f + expf(-zo));
      float cc = sf * cbuf[b * HID + j] + si * tanhf(zg);
      float hh = so * tanhf(cc);
      cbuf[b * HID + j] = cc;
      hB[b * HID + j] = f2bf(hh);
      hiddens [(size_t)(t * BATCH + b) * HID + j] = hh;
      hiddensB[(size_t)(t * BATCH + b) * HID + j] = f2bf(hh);
    }
    grid_barrier(sync_count, sync_flag, 2 * t + 1);  // all writes of hB visible
  }
}

// ---------------------------------------------------------------------------
// Causal attention scores + softmax (incl. sentinel). One WG per (t,b).
// z[t,j,b] = <hiddens[j,b,:], queries[t,b,:]>, j<=t; s = <queries, sentinel>.
// ---------------------------------------------------------------------------
__global__ void __launch_bounds__(256) attn_kernel(const float* __restrict__ hiddens,
                                                   const float* __restrict__ queries,
                                                   const float* __restrict__ sentinel,
                                                   float* __restrict__ a_ptr,
                                                   float* __restrict__ a_sent) {
  const int t = blockIdx.x >> 5;
  const int b = blockIdx.x & 31;
  __shared__ float q[HID];
  __shared__ float sc[L_SEQ + 1];
  __shared__ float ms[2];
  const int tid = threadIdx.x;
  const int wave = tid >> 5, lane = tid & 31;

  for (int i = tid; i < HID; i += 256) q[i] = queries[(size_t)(t * BATCH + b) * HID + i];
  __syncthreads();

  for (int j = wave; j <= t; j += 8) {
    const float* hr = hiddens + (size_t)(j * BATCH + b) * HID;
    float s = 0.f;
    for (int i = lane; i < HID; i += 32) s += hr[i] * q[i];
    for (int off = 16; off > 0; off >>= 1) s += __shfl_xor(s, off, 32);
    if (lane == 0) sc[j] = s;
  }
  if (wave == 0) {
    float s = 0.f;
    for (int i = lane; i < HID; i += 32) s += q[i] * sentinel[i];
    for (int off = 16; off > 0; off >>= 1) s += __shfl_xor(s, off, 32);
    if (lane == 0) sc[L_SEQ] = s;
  }
  __syncthreads();

  if (tid == 0) {
    float m = sc[L_SEQ];
    for (int j = 0; j <= t; ++j) m = fmaxf(m, sc[j]);
    float sum = expf(sc[L_SEQ] - m);
    for (int j = 0; j <= t; ++j) sum += expf(sc[j] - m);
    ms[0] = m; ms[1] = sum;
  }
  __syncthreads();
  const float m = ms[0], inv = 1.f / ms[1];
  for (int j = tid; j <= t; j += 256)
    a_ptr[(size_t)(t * L_SEQ + j) * BATCH + b] = expf(sc[j] - m) * inv;
  if (tid == 0) a_sent[t * BATCH + b] = expf(sc[L_SEQ] - m) * inv;
}

// ---------------------------------------------------------------------------
// Per-row: softmax(logits)*a_sent + pointer scatter (serialized, deterministic)
// then log, all in place in d_out. One WG per row.
// ---------------------------------------------------------------------------
__global__ void __launch_bounds__(256) finalize_kernel(float* __restrict__ out,
                                                       const int* __restrict__ tokens,
                                                       const float* __restrict__ a_ptr,
                                                       const float* __restrict__ a_sent) {
  const int r = blockIdx.x;        // r = t*B + b
  const int t = r >> 5;
  const int b = r & 31;
  float* row = out + (size_t)r * VOCAB;
  __shared__ float red[256];
  __shared__ float ms[2];
  const int tid = threadIdx.x;

  float m = -3.4e38f;
  for (int c = tid; c < VOCAB; c += 256) m = fmaxf(m, row[c]);
  red[tid] = m; __syncthreads();
  if (tid == 0) { float mm = red[0]; for (int i = 1; i < 256; ++i) mm = fmaxf(mm, red[i]); ms[0] = mm; }
  __syncthreads();
  m = ms[0];

  float s = 0.f;
  for (int c = tid; c < VOCAB; c += 256) s += expf(row[c] - m);
  __syncthreads();
  red[tid] = s; __syncthreads();
  if (tid == 0) { float ss = 0.f; for (int i = 0; i < 256; ++i) ss += red[i]; ms[1] = ss; }
  __syncthreads();

  const float scale = a_sent[r] / ms[1];
  for (int c = tid; c < VOCAB; c += 256) row[c] = expf(row[c] - m) * scale;
  __threadfence_block();
  __syncthreads();

  if (tid == 0) {                   // deterministic serialized pointer add
    for (int j = 0; j <= t; ++j) {
      int tok = tokens[j * BATCH + b];
      row[tok] += a_ptr[(size_t)(t * L_SEQ + j) * BATCH + b];
    }
  }
  __threadfence_block();
  __syncthreads();

  for (int c = tid; c < VOCAB; c += 256) row[c] = logf(row[c]);
}

// ---------------------------------------------------------------------------
// Launch
// ---------------------------------------------------------------------------
extern "C" void kernel_launch(void* const* d_in, const int* in_sizes, int n_in,
                              void* d_out, int out_size, void* d_ws, size_t ws_size,
                              hipStream_t stream) {
  (void)in_sizes; (void)n_in; (void)out_size; (void)ws_size;
  const int*   tokens      = (const int*)  d_in[0];
  const float* embed_table = (const float*)d_in[1];
  const float* W_ih        = (const float*)d_in[2];
  const float* W_hh        = (const float*)d_in[3];
  const float* b_ih        = (const float*)d_in[4];
  const float* b_hh        = (const float*)d_in[5];
  const float* W1          = (const float*)d_in[6];
  const float* b1          = (const float*)d_in[7];
  const float* W2          = (const float*)d_in[8];
  const float* b2          = (const float*)d_in[9];
  const float* sentinel    = (const float*)d_in[10];
  float* out = (float*)d_out;

  char* ws = (char*)d_ws;
  size_t off = 0;
  auto take = [&](size_t bytes) -> void* {
    void* p = ws + off;
    off = (off + bytes + 255) & ~(size_t)255;
    return p;
  };
  u16*   embB    = (u16*)  take((size_t)ROWS * HID * 2);
  u16*   WihB    = (u16*)  take((size_t)GATES * HID * 2);
  u16*   WhhB    = (u16*)  take((size_t)GATES * HID * 2);
  u16*   W1B     = (u16*)  take((size_t)VOCAB * HID * 2);
  u16*   W2B     = (u16*)  take((size_t)HID * HID * 2);
  float* biasC   = (float*)take((size_t)GATES * 4);
  float* gatesx  = (float*)take((size_t)ROWS * GATES * 4);
  float* hiddens = (float*)take((size_t)ROWS * HID * 4);
  u16*   hiddB   = (u16*)  take((size_t)ROWS * HID * 2);
  float* queries = (float*)take((size_t)ROWS * HID * 4);
  u16*   hB      = (u16*)  take((size_t)BATCH * HID * 2);
  float* cbuf    = (float*)take((size_t)BATCH * HID * 4);
  float* aptr    = (float*)take((size_t)L_SEQ * L_SEQ * BATCH * 4);
  float* asent   = (float*)take((size_t)ROWS * 4);
  int*   sync_   = (int*)  take(512);

  // allow the 280 KB dynamic-LDS allocation (CDNA5: 320 KB per WGP)
  hipFuncSetAttribute(reinterpret_cast<const void*>(lstm_kernel),
                      hipFuncAttributeMaxDynamicSharedMemorySize, LSTM_LDS_BYTES);

  // zero-init recurrent state + barrier state (graph-capture-safe memsets)
  hipMemsetAsync(hB,    0, (size_t)BATCH * HID * 2, stream);
  hipMemsetAsync(cbuf,  0, (size_t)BATCH * HID * 4, stream);
  hipMemsetAsync(sync_, 0, 512, stream);

  embed_gather_kernel<<<ROWS, 256, 0, stream>>>(tokens, embed_table, embB);
  f32_to_bf16_kernel<<<(GATES * HID + 255) / 256, 256, 0, stream>>>(W_ih, WihB, GATES * HID);
  f32_to_bf16_kernel<<<(GATES * HID + 255) / 256, 256, 0, stream>>>(W_hh, WhhB, GATES * HID);
  f32_to_bf16_kernel<<<(VOCAB * HID + 255) / 256, 256, 0, stream>>>(W1, W1B, VOCAB * HID);
  f32_to_bf16_kernel<<<(HID * HID + 255) / 256, 256, 0, stream>>>(W2, W2B, HID * HID);
  bias_combine_kernel<<<(GATES + 255) / 256, 256, 0, stream>>>(b_ih, b_hh, biasC, GATES);

  // gates_x = emb @ W_ih^T + (b_ih + b_hh)
  gemm_bf16_kernel<0><<<dim3(GATES / 64, ROWS / 64), 256, 0, stream>>>(
      embB, WihB, biasC, gatesx, ROWS, GATES, HID);

  // sequential recurrence, cooperative across 32 WGPs, weights LDS-resident
  lstm_kernel<<<LSTM_WGS, 256, LSTM_LDS_BYTES, stream>>>(gatesx, WhhB, hiddens, hiddB,
                                                         hB, cbuf, sync_, sync_ + 64);

  // queries = tanh(hiddens @ W2^T + b2)
  gemm_bf16_kernel<1><<<dim3(HID / 64, ROWS / 64), 256, 0, stream>>>(
      hiddB, W2B, b2, queries, ROWS, HID, HID);

  attn_kernel<<<ROWS, 256, 0, stream>>>(hiddens, queries, sentinel, aptr, asent);

  // vocab logits straight into d_out
  gemm_bf16_kernel<0><<<dim3(VOCAB / 64, ROWS / 64), 256, 0, stream>>>(
      hiddB, W1B, b1, out, ROWS, VOCAB, HID);

  finalize_kernel<<<ROWS, 256, 0, stream>>>(out, tokens, aptr, asent);
}
